// DynamicFilterNetwork_20787641713149
// MI455X (gfx1250) — matrix operand order repaired and checked by reading, not compiled, not run on hardware
//
#include <hip/hip_runtime.h>
#include <hip/hip_bf16.h>
#include <cstdint>

typedef _Float16 half_t;
typedef __attribute__((ext_vector_type(16))) _Float16     v16h;
typedef __attribute__((ext_vector_type(8)))  float        v8f;
typedef __attribute__((ext_vector_type(8)))  unsigned int v8u;

#define HH 256
#define WW 256
#define BB 8
#define PH 258   // padded height (1-px zero halo)
#define PW 258   // padded width
static constexpr long NPIX  = (long)BB * HH * WW;   // 524288
static constexpr long NPPIX = (long)BB * PH * PW;   // 532512

static __device__ __forceinline__ unsigned pack2(float a, float b) {
    _Float16 ha = (_Float16)a, hb = (_Float16)b;
    unsigned short ua = __builtin_bit_cast(unsigned short, ha);
    unsigned short ub = __builtin_bit_cast(unsigned short, hb);
    return (unsigned)ua | ((unsigned)ub << 16);
}

// ---------------------------------------------------------------------------
// Vectorized zero fill (for the halo'd activation buffers).
// ---------------------------------------------------------------------------
__global__ __launch_bounds__(256) void zero_kernel(uint4* __restrict__ p, long n4) {
    long i = (long)blockIdx.x * 256 + threadIdx.x;
    if (i < n4) p[i] = make_uint4(0u, 0u, 0u, 0u);
}

// ---------------------------------------------------------------------------
// Pack OIHW f32 weights -> per-lane-contiguous A-matrix f16 pairs:
//   apack[(((mt*NC + kc)*32 + lane) * 8) + v]
// holds K pair (k, k+1), k = kc*32 + kk, kk = (v>>2)*16 + hf*8 + (v&3)*2,
// row M = mt*16 + (lane&15), hf = lane>>4.  K index k = r*CIN + c (r=ky*3+kx).
// Each wave lane then fetches its whole A chunk with one 32B load.
// ---------------------------------------------------------------------------
__global__ void pack_w_kernel(const float* __restrict__ w, unsigned* __restrict__ apack,
                              int COUT, int CIN, int MT) {
    const int NC = CIN * 9 / 32;
    const long total = (long)MT * NC * 32 * 8;
    long id = (long)blockIdx.x * blockDim.x + threadIdx.x;
    if (id >= total) return;
    int v    = (int)(id & 7);
    int lane = (int)((id >> 3) & 31);
    int kc   = (int)((id >> 8) % NC);
    int mt   = (int)(id / (256L * NC));
    int m = lane & 15, hf = lane >> 4;
    int kk = ((v >> 2) << 4) + (hf << 3) + ((v & 3) << 1);
    int k0 = kc * 32 + kk, k1 = k0 + 1;
    int mg = mt * 16 + m;
    float v0 = 0.f, v1 = 0.f;
    if (mg < COUT) {
        int r0 = k0 / CIN, c0 = k0 % CIN;
        int r1 = k1 / CIN, c1 = k1 % CIN;
        v0 = w[((mg * CIN + c0) * 3 + r0 / 3) * 3 + (r0 % 3)];
        v1 = w[((mg * CIN + c1) * 3 + r1 / 3) * 3 + (r1 % 3)];
    }
    apack[id] = pack2(v0, v1);
}

// ---------------------------------------------------------------------------
// conv1: Cin=4 (rgb+depth, NCHW f32) -> 64ch, scalar f32, ReLU,
// f16 NHWC output into the padded (halo) buffer.
// ---------------------------------------------------------------------------
__global__ __launch_bounds__(256) void conv1_kernel(
    const float* __restrict__ rgb, const float* __restrict__ depth,
    const float* __restrict__ w1, const float* __restrict__ b1,
    half_t* __restrict__ out) {
    long pix = (long)blockIdx.x * 256 + threadIdx.x;
    if (pix >= NPIX) return;
    int b   = (int)(pix / (HH * WW));
    int rem = (int)(pix % (HH * WW));
    int y = rem / WW, x = rem % WW;
    float acc[64];
#pragma unroll
    for (int o = 0; o < 64; ++o) acc[o] = 0.f;
    for (int ky = 0; ky < 3; ++ky)
        for (int kx = 0; kx < 3; ++kx) {
            int yy = y + ky - 1, xx = x + kx - 1;
            bool inb = ((unsigned)yy < HH) && ((unsigned)xx < WW);
            int yyc = inb ? yy : 0, xxc = inb ? xx : 0;
            long base = ((long)b * 3) * HH * WW + (long)yyc * WW + xxc;
            float r0 = rgb[base];
            float r1 = rgb[base + (long)HH * WW];
            float r2 = rgb[base + 2L * HH * WW];
            float dd = depth[((long)b * HH + yyc) * WW + xxc];
            float xin[4];
            xin[0] = inb ? r0 : 0.f;
            xin[1] = inb ? r1 : 0.f;
            xin[2] = inb ? r2 : 0.f;
            xin[3] = inb ? dd : 0.f;
            const float* wp = w1 + (ky * 3 + kx);
            for (int c = 0; c < 4; ++c) {
                float xv = xin[c];
                const float* wc = wp + c * 9;
#pragma unroll
                for (int o = 0; o < 64; ++o) acc[o] += xv * wc[o * 36];
            }
        }
    long ppix = ((long)b * PH + (y + 1)) * PW + (x + 1);
    unsigned* op = (unsigned*)out + ppix * 32;
#pragma unroll
    for (int o = 0; o < 64; o += 2) {
        float v0 = acc[o] + b1[o];         v0 = v0 > 0.f ? v0 : 0.f;
        float v1 = acc[o + 1] + b1[o + 1]; v1 = v1 > 0.f ? v1 : 0.f;
        op[o >> 1] = pack2(v0, v1);
    }
}

// ---------------------------------------------------------------------------
// WMMA implicit-GEMM 3x3 SAME conv over zero-halo'd NHWC f16 activations.
// 128 threads = 4 waves; each wave owns a 16-outchannel x 16-pixel tile;
// block covers 64 consecutive x pixels.
//   M = out channels (tile mt), N = pixels, K = CIN*9 in 32-chunks (one 3x3
//   tap per chunk since CIN % 32 == 0).
// Per chunk: one 32B A load + two 16B B loads + one v_wmma_f32_16x16x32_f16.
// No bounds logic at all (halo supplies the zeros).
// Epilogue: bias+ReLU+f16 NHWC b128 store, or (SOFTMAX) 11-ch softmax -> f32.
// ---------------------------------------------------------------------------
template <int CIN, bool SOFTMAX>
__global__ __launch_bounds__(128) void conv3_wmma_kernel(
    const half_t* __restrict__ in, const unsigned* __restrict__ apack,
    const float* __restrict__ bias, half_t* __restrict__ out_f16,
    float* __restrict__ out_sm, int COUT, int MT) {
    constexpr int K  = CIN * 9;
    constexpr int NC = K / 32;
    const int lane = threadIdx.x & 31;
    const int wav  = threadIdx.x >> 5;
    const int n    = lane & 15;   // B/D column, also A row (M)
    const int hf   = lane >> 4;   // half-wave selector
    const int x0   = blockIdx.x * 64;
    const int y    = blockIdx.y;
    const int b    = blockIdx.z / MT;
    const int mt   = blockIdx.z % MT;
    const int xc   = x0 + wav * 16 + n;

    const v8u* abase = (const v8u*)apack + (size_t)(mt * NC) * 32 + lane;
    const unsigned* inu = (const unsigned*)in;
    // top-left tap of this output pixel in the padded buffer
    const long pbase = ((long)b * PH + y) * PW + xc;

    v8f acc = {};
#pragma unroll 2
    for (int kc = 0; kc < NC; ++kc) {
        const int kbase = kc * 32;
        const int r     = kbase / CIN;     // 3x3 tap id (chunk stays in one tap)
        const int cbase = kbase % CIN;
        const long pixi = pbase + (r / 3) * PW + (r % 3);
        // B operand: lane=column n, VGPR j holds K = (j>>2)*16 + hf*8 + 2(j&3), +1
        const unsigned* bp = inu + pixi * (CIN / 2) + (cbase >> 1) + hf * 4;
        uint4 b0 = *(const uint4*)bp;
        uint4 b1 = *(const uint4*)(bp + 8);
        v8u bu = {b0.x, b0.y, b0.z, b0.w, b1.x, b1.y, b1.z, b1.w};
        // A operand: one contiguous 32B load per lane (pre-shuffled layout)
        v8u au = abase[(size_t)kc * 32];
        v16h a  = __builtin_bit_cast(v16h, au);
        v16h bm = __builtin_bit_cast(v16h, bu);
        acc = __builtin_amdgcn_wmma_f32_16x16x32_f16(
            /*neg_a=*/false, a, /*neg_b=*/false, bm,
            /*c_mod=*/(short)0, acc, /*reuse_a=*/false, /*reuse_b=*/false);
    }

    if constexpr (!SOFTMAX) {
        // D layout: lane=column n, VGPR v -> channel m = v + 8*hf
        unsigned* outu = (unsigned*)out_f16;
        long ppix = ((long)b * PH + (y + 1)) * PW + (xc + 1);
        unsigned* op = outu + ppix * (COUT >> 1) + mt * 8 + hf * 4;
        uint4 st;
        {
            int m0 = mt * 16 + 8 * hf;
            float a0 = acc[0] + bias[m0 + 0]; a0 = a0 > 0.f ? a0 : 0.f;
            float a1 = acc[1] + bias[m0 + 1]; a1 = a1 > 0.f ? a1 : 0.f;
            float a2 = acc[2] + bias[m0 + 2]; a2 = a2 > 0.f ? a2 : 0.f;
            float a3 = acc[3] + bias[m0 + 3]; a3 = a3 > 0.f ? a3 : 0.f;
            float a4 = acc[4] + bias[m0 + 4]; a4 = a4 > 0.f ? a4 : 0.f;
            float a5 = acc[5] + bias[m0 + 5]; a5 = a5 > 0.f ? a5 : 0.f;
            float a6 = acc[6] + bias[m0 + 6]; a6 = a6 > 0.f ? a6 : 0.f;
            float a7 = acc[7] + bias[m0 + 7]; a7 = a7 > 0.f ? a7 : 0.f;
            st.x = pack2(a0, a1); st.y = pack2(a2, a3);
            st.z = pack2(a4, a5); st.w = pack2(a6, a7);
        }
        *(uint4*)op = st;
    } else {
        __shared__ float sm[4][16][17];
#pragma unroll
        for (int v = 0; v < 8; ++v) sm[wav][v + 8 * hf][n] = acc[v];
        __syncthreads();
        int tid = threadIdx.x;
        if (tid < 64) {
            int wv = tid >> 4, nn = tid & 15;
            int px = x0 + wv * 16 + nn;
            float vals[11];
            float mx = -1e30f;
#pragma unroll
            for (int i = 0; i < 11; ++i) {
                vals[i] = sm[wv][i][nn] + bias[i];
                mx = fmaxf(mx, vals[i]);
            }
            float s = 0.f;
#pragma unroll
            for (int i = 0; i < 11; ++i) { vals[i] = __expf(vals[i] - mx); s += vals[i]; }
            float inv = 1.f / s;
#pragma unroll
            for (int i = 0; i < 11; ++i)
                out_sm[(((long)b * 11 + i) * HH + y) * WW + px] = vals[i] * inv;
        }
    }
}

// ---------------------------------------------------------------------------
// Separable dynamic filter: horizontal pass (rgb * kh -> hbuf).
// ---------------------------------------------------------------------------
__global__ __launch_bounds__(256) void hfilter_kernel(
    const float* __restrict__ rgb, const float* __restrict__ kh,
    float* __restrict__ hb) {
    long pix = (long)blockIdx.x * 256 + threadIdx.x;
    if (pix >= NPIX) return;
    int b   = (int)(pix / (HH * WW));
    int rem = (int)(pix % (HH * WW));
    int y = rem / WW, x = rem % WW;
    float k[11];
#pragma unroll
    for (int i = 0; i < 11; ++i) k[i] = kh[(((long)b * 11 + i) * HH + y) * WW + x];
#pragma unroll
    for (int c = 0; c < 3; ++c) {
        const float* rp = rgb + (((long)b * 3 + c) * HH + y) * WW;
        float s = 0.f;
#pragma unroll
        for (int i = 0; i < 11; ++i) {
            int xx = x + i - 5;
            float rv = rp[(unsigned)xx < WW ? xx : x];   // clamped load
            s += ((unsigned)xx < WW) ? k[i] * rv : 0.f;
        }
        hb[(((long)b * 3 + c) * HH + y) * WW + x] = s;
    }
}

// ---------------------------------------------------------------------------
// Vertical pass + depth-mask blend: writes blurred, final, mask.
// ---------------------------------------------------------------------------
__global__ __launch_bounds__(256) void vfilter_blend_kernel(
    const float* __restrict__ rgb, const float* __restrict__ depth,
    const float* __restrict__ kv, const float* __restrict__ hb,
    float* __restrict__ final_o, float* __restrict__ blur_o,
    float* __restrict__ mask_o) {
    long pix = (long)blockIdx.x * 256 + threadIdx.x;
    if (pix >= NPIX) return;
    int b   = (int)(pix / (HH * WW));
    int rem = (int)(pix % (HH * WW));
    int y = rem / WW, x = rem % WW;
    float k[11];
#pragma unroll
    for (int i = 0; i < 11; ++i) k[i] = kv[(((long)b * 11 + i) * HH + y) * WW + x];
    float mask = depth[pix] > 0.2f ? 1.f : 0.f;
    mask_o[pix] = mask;
#pragma unroll
    for (int c = 0; c < 3; ++c) {
        const float* hp = hb + (((long)b * 3 + c) * HH) * WW + x;
        float s = 0.f;
#pragma unroll
        for (int i = 0; i < 11; ++i) {
            int yy = y + i - 5;
            float hv = hp[(long)((unsigned)yy < HH ? yy : y) * WW];  // clamped load
            s += ((unsigned)yy < HH) ? k[i] * hv : 0.f;
        }
        long oi = (((long)b * 3 + c) * HH + y) * WW + x;
        blur_o[oi]  = s;
        final_o[oi] = mask * rgb[oi] + (1.f - mask) * s;
    }
}

// ---------------------------------------------------------------------------
extern "C" void kernel_launch(void* const* d_in, const int* in_sizes, int n_in,
                              void* d_out, int out_size, void* d_ws, size_t ws_size,
                              hipStream_t stream) {
    const float* rgb   = (const float*)d_in[0];
    const float* depth = (const float*)d_in[1];
    const float* w1 = (const float*)d_in[2];  const float* b1 = (const float*)d_in[3];
    const float* w2 = (const float*)d_in[4];  const float* b2 = (const float*)d_in[5];
    const float* w3 = (const float*)d_in[6];  const float* b3 = (const float*)d_in[7];
    const float* w4 = (const float*)d_in[8];  const float* b4 = (const float*)d_in[9];
    const float* w5 = (const float*)d_in[10]; const float* b5 = (const float*)d_in[11];
    const float* wh = (const float*)d_in[12]; const float* bh = (const float*)d_in[13];
    const float* wv = (const float*)d_in[14]; const float* bv = (const float*)d_in[15];

    float* outF    = (float*)d_out;
    float* o_final = outF;
    float* o_blur  = outF + 3 * NPIX;
    float* o_kh    = outF + 6 * NPIX;
    float* o_kv    = outF + 6 * NPIX + 11 * NPIX;
    float* o_mask  = outF + 6 * NPIX + 22 * NPIX;

    char* ws = (char*)d_ws;
    size_t off = 0;
    auto alloc = [&](size_t bytes) -> void* {
        void* p = ws + off;
        off += (bytes + 255) & ~(size_t)255;
        return p;
    };
    const size_t actBytes = (size_t)NPPIX * 128 * 2;           // 136,323,072 (256-aligned)
    half_t*   actA = (half_t*)alloc(actBytes);                 // ping (<=128 ch, halo'd)
    half_t*   actB = (half_t*)alloc(actBytes);                 // pong (<=128 ch, halo'd)
    unsigned* ap2  = (unsigned*)alloc((size_t)4 * 18 * 256 * 4);
    unsigned* ap3  = (unsigned*)alloc((size_t)8 * 18 * 256 * 4);
    unsigned* ap4  = (unsigned*)alloc((size_t)8 * 36 * 256 * 4);
    unsigned* ap5  = (unsigned*)alloc((size_t)4 * 36 * 256 * 4);
    unsigned* aph  = (unsigned*)alloc((size_t)1 * 18 * 256 * 4);
    unsigned* apv  = (unsigned*)alloc((size_t)1 * 18 * 256 * 4);
    float* hbuf = (float*)actB;  // actB is dead by the time the blur runs

    // Zero both activation buffers (supplies the zero halos; interior is
    // overwritten by the convs).  actA/actB are contiguous -> one launch.
    {
        long n4 = (long)(2 * actBytes / 16);
        zero_kernel<<<(unsigned)((n4 + 255) / 256), 256, 0, stream>>>((uint4*)actA, n4);
    }

    auto packs = [&](const float* w, unsigned* ap, int cout, int cin, int mtl) {
        long total = (long)mtl * (cin * 9 / 32) * 256;
        pack_w_kernel<<<(unsigned)((total + 255) / 256), 256, 0, stream>>>(w, ap, cout, cin, mtl);
    };
    packs(w2, ap2, 64, 64, 4);
    packs(w3, ap3, 128, 64, 8);
    packs(w4, ap4, 128, 128, 8);
    packs(w5, ap5, 64, 128, 4);
    packs(wh, aph, 11, 64, 1);
    packs(wv, apv, 11, 64, 1);

    conv1_kernel<<<(unsigned)(NPIX / 256), 256, 0, stream>>>(rgb, depth, w1, b1, actA);

    dim3 blk(128);
    conv3_wmma_kernel<64,  false><<<dim3(4, 256, 8 * 4), blk, 0, stream>>>(actA, ap2, b2, actB, nullptr, 64, 4);
    conv3_wmma_kernel<64,  false><<<dim3(4, 256, 8 * 8), blk, 0, stream>>>(actB, ap3, b3, actA, nullptr, 128, 8);
    conv3_wmma_kernel<128, false><<<dim3(4, 256, 8 * 8), blk, 0, stream>>>(actA, ap4, b4, actB, nullptr, 128, 8);
    conv3_wmma_kernel<128, false><<<dim3(4, 256, 8 * 4), blk, 0, stream>>>(actB, ap5, b5, actA, nullptr, 64, 4);
    conv3_wmma_kernel<64,  true ><<<dim3(4, 256, 8),     blk, 0, stream>>>(actA, aph, bh, nullptr, o_kh, 16, 1);
    conv3_wmma_kernel<64,  true ><<<dim3(4, 256, 8),     blk, 0, stream>>>(actA, apv, bv, nullptr, o_kv, 16, 1);

    hfilter_kernel<<<(unsigned)(NPIX / 256), 256, 0, stream>>>(rgb, o_kh, hbuf);
    vfilter_blend_kernel<<<(unsigned)(NPIX / 256), 256, 0, stream>>>(
        rgb, depth, o_kv, hbuf, o_final, o_blur, o_mask);
}
